// LTCPolicy_20564303413880
// MI455X (gfx1250) — matrix operand order repaired and checked by reading, not compiled, not run on hardware
//
#include <hip/hip_runtime.h>

typedef _Float16 half_t;
typedef __attribute__((ext_vector_type(16))) _Float16 v16h;
typedef __attribute__((ext_vector_type(8)))  _Float16 v8h;
typedef __attribute__((ext_vector_type(8)))  float    v8f;

union V16 { v16h v; v8h h[2]; };

#define B_SZ 512
#define T_SZ 1024

// ---- workspace (d_ws) byte offsets: f16 fragment arrays (pre-swizzled B layout) ----
// K is augmented to 96 for GEMM1/GEMM2: rows 64..66 = input (K=3) weights, row 67 = bias,
// rows 68..95 = 0. The matching aux A fragment carries [obs0,obs1,obs2,1] at k=64..67.
#define WS_WHF 0        // lstm_wh|lstm_wi|lstm_bi aug 96x256 : 3kt*16nt*32lane*16h*2B = 49152 B
#define WS_BBF 49152    // bb_wh|bb_w[:3]|bb_b aug 96x128     : 3*8  -> 24576 B
#define WS_FFF 73728    // ff1|ff2|(ta+tb) 128x192            : 4*12 -> 49152 B
#define WS_PJF 122880   // proj 64x16                         : 2*1  -> 2048 B
#define WS_V2F 124928   // v2   64x64                         : 2*4  -> 8192 B
// total ws bytes needed: 133120

// ---- main-kernel LDS layout (frag block mirrors ws[0..124928)) ----
#define WHF_OFF 0
#define BBF_OFF 49152
#define FFF_OFF 73728
#define PJF_OFF 122880
#define SCR_OFF 124928   // shared per-tile scratch: hL_s(2048) + hN_s(2048) + bb_s(4096)
#define LDS_TOTAL (SCR_OFF + 8192)   // 133120 B (< 320KB WGP LDS)

__device__ __forceinline__ v16h ld_frag(const half_t* p) {
    V16 u;
    u.h[0] = *(const v8h*)(p);
    u.h[1] = *(const v8h*)(p + 8);
    return u.v;
}
// A-layout fragment from row-major f16 matrix in LDS (stride K halfs)
__device__ __forceinline__ v16h ld_afrag(const half_t* hs, int K, int kt, int m, int hsel) {
    V16 u;
    const half_t* p = hs + m*K + kt*32 + hsel*16;
    u.h[0] = *(const v8h*)(p);        // K = kt*32 + hsel*16 + [0..7]
    u.h[1] = *(const v8h*)(p + 8);    // K = kt*32 + hsel*16 + 8 + [0..7]
    return u.v;
}
__device__ __forceinline__ v8f wmma16(v16h a, v16h b, v8f c) {
    return __builtin_amdgcn_wmma_f32_16x16x32_f16(false, a, false, b, (short)0, c, false, false);
}
__device__ __forceinline__ v8f vsplat(float x) {
    v8f r;
#pragma unroll
    for (int j = 0; j < 8; ++j) r[j] = x;
    return r;
}
__device__ __forceinline__ v16h hzero() {
    V16 u;
#pragma unroll
    for (int j = 0; j < 16; ++j) u.v[j] = (half_t)0.0f;
    return u.v;
}
// CDNA5 native v_tanh_f32 (confirmed lowering in round 2); fallback kept for portability.
__device__ __forceinline__ float ftanh(float x) {
#if __has_builtin(__builtin_amdgcn_tanhf)
    return __builtin_amdgcn_tanhf(x);
#else
    x = fminf(fmaxf(x, -15.0f), 15.0f);
    float e = __expf(2.0f * x);
    return (e - 1.0f) * __builtin_amdgcn_rcpf(e + 1.0f);
#endif
}
__device__ __forceinline__ float fsigm(float x) {
    return fmaf(0.5f, ftanh(0.5f * x), 0.5f);
}

// ===================== prep: f32 weights -> f16 swizzled B fragments =====================
// B-fragment layout (16x16x32 f16): lane l holds N=l&15, K=(l>>4)*16 + j, j=0..15,
// stored lane-major: frag[(kt*NT+nt)*32 + lane][16]
__global__ void prep_kernel(const float* lstm_wh, const float* lstm_wi, const float* lstm_bi,
                            const float* bb_w, const float* bb_b,
                            const float* ff1_w, const float* ff2_w,
                            const float* ta_w, const float* tb_w,
                            const float* proj_w, const float* v2_w,
                            half_t* ws)
{
    int tid = blockIdx.x * blockDim.x + threadIdx.x;
    int stride = gridDim.x * blockDim.x;

    for (int e = tid; e < 3*16*512; e += stride) {            // Wh aug: KT=3 NT=16 N=256
        int j = e & 15, lane = (e >> 4) & 31, f = e >> 9;
        int kt = f / 16, nt = f % 16;
        int k = kt*32 + (lane >> 4)*16 + j, n = nt*16 + (lane & 15);
        float v;
        if (k < 64)       v = lstm_wh[k*256 + n];
        else if (k < 67)  v = lstm_wi[(k - 64)*256 + n];
        else if (k == 67) v = lstm_bi[n];
        else              v = 0.0f;
        ws[WS_WHF/2 + e] = (half_t)v;
    }
    for (int e = tid; e < 3*8*512; e += stride) {             // bb aug: KT=3 NT=8 N=128
        int j = e & 15, lane = (e >> 4) & 31, f = e >> 9;
        int kt = f / 8, nt = f % 8;
        int k = kt*32 + (lane >> 4)*16 + j, n = nt*16 + (lane & 15);
        float v;
        if (k < 64)       v = bb_w[(3 + k)*128 + n];
        else if (k < 67)  v = bb_w[(k - 64)*128 + n];
        else if (k == 67) v = bb_b[n];
        else              v = 0.0f;
        ws[WS_BBF/2 + e] = (half_t)v;
    }
    for (int e = tid; e < 4*12*512; e += stride) {            // ffcat KT=4 NT=12 N=192
        int j = e & 15, lane = (e >> 4) & 31, f = e >> 9;
        int kt = f / 12, nt = f % 12;
        int k = kt*32 + (lane >> 4)*16 + j, n = nt*16 + (lane & 15);
        float v;
        if (n < 64)       v = ff1_w[k*64 + n];
        else if (n < 128) v = ff2_w[k*64 + (n - 64)];
        else              v = ta_w[k*64 + (n - 128)] + tb_w[k*64 + (n - 128)];
        ws[WS_FFF/2 + e] = (half_t)v;
    }
    for (int e = tid; e < 2*1*512; e += stride) {             // proj KT=2 NT=1 N=16
        int j = e & 15, lane = (e >> 4) & 31, f = e >> 9;
        int k = f*32 + (lane >> 4)*16 + j, n = (lane & 15);
        ws[WS_PJF/2 + e] = (half_t)proj_w[k*16 + n];
    }
    for (int e = tid; e < 2*4*512; e += stride) {             // v2 KT=2 NT=4 N=64
        int j = e & 15, lane = (e >> 4) & 31, f = e >> 9;
        int kt = f >> 2, nt = f & 3;
        int k = kt*32 + (lane >> 4)*16 + j, n = nt*16 + (lane & 15);
        ws[WS_V2F/2 + e] = (half_t)v2_w[k*64 + n];
    }
}

// ============ main recurrent kernel: 1 workgroup (4 waves) per 16-row batch tile ============
// Wave w owns unit quarter u in [16w,16w+16): gate n-tiles {w,4+w,8+w,12+w}, bb n-tiles
// {2w,2w+1}, ff n-tile w. LSTM elementwise stays wave-local; h/bb are exchanged through
// LDS with 3 workgroup split-barriers per step. hL_s/hN_s are separate buffers so the
// next step's h_lstm store can never race a lagging wave's h_new A-fragment load.
__global__ __launch_bounds__(128) void ltc_kernel(
    const float* __restrict__ obs, const float* __restrict__ rms_mean, const float* __restrict__ rms_var,
    const float* __restrict__ ff1_b, const float* __restrict__ ff2_b,
    const float* __restrict__ ta_b, const float* __restrict__ tb_b,
    const float* __restrict__ proj_b, const float* __restrict__ am_w, const float* __restrict__ am_b,
    const half_t* __restrict__ wsf,
    float* __restrict__ out_mean, float* __restrict__ out_hT, float* __restrict__ out_cT)
{
    extern __shared__ char smem_raw[];
    half_t* sm = (half_t*)smem_raw;
    const int tid  = threadIdx.x;
    const int lane = tid & 31;
    const int wv   = tid >> 5;

    // stage weight fragments ws[0..124928) -> LDS (identical offsets)
    {
        const uint4* src = (const uint4*)wsf;
        uint4* dst = (uint4*)smem_raw;
        for (int i = tid; i < SCR_OFF/16; i += blockDim.x) dst[i] = src[i];
    }
    __syncthreads();

    half_t* hL_s = (half_t*)(smem_raw + SCR_OFF);          // 16x64  f16 (h_lstm exchange)
    half_t* hN_s = (half_t*)(smem_raw + SCR_OFF + 2048);   // 16x64  f16 (h_new exchange)
    half_t* bb_s = (half_t*)(smem_raw + SCR_OFF + 4096);   // 16x128 f16

    const int brow = blockIdx.x * 16;
    const int col  = lane & 15;
    const int hsel = lane >> 4;
    const int m    = lane & 15;

    const float nm0 = rms_mean[0], nm1 = rms_mean[1], nm2 = rms_mean[2];
    const float is0 = 1.0f / (sqrtf(rms_var[0]) + 1e-8f);
    const float is1 = 1.0f / (sqrtf(rms_var[1]) + 1e-8f);
    const float is2 = 1.0f / (sqrtf(rms_var[2]) + 1e-8f);
    const float amb = am_b[0];

    // loop-invariant biases for this wave's ff n-tile and for proj
    const int n3 = wv*16 + col;
    const float fb1_r = ff1_b[n3];
    const float fb2_r = ff2_b[n3];
    const float fbt_r = ta_b[n3] + tb_b[n3];
    const float pjb_r = proj_b[col];
    const float aw    = am_w[col];

    v8f c = vsplat(0.0f), ti = vsplat(0.0f), hl, hn;
    v16h ha0 = hzero(), ha1 = hzero();   // h_new in A layout (zero initial state)

    const float* myobs = obs + (size_t)(brow + m) * T_SZ * 3;

#pragma unroll 1
    for (int t = 0; t < T_SZ; ++t) {
        // ---- aux A fragment: k=64..66 normalized obs, k=67 = 1.0 (weight rows >=68 are 0) ----
        V16 ax;
#pragma unroll
        for (int j = 0; j < 16; ++j) ax.v[j] = (half_t)0.0f;
        if (hsel == 0) {
            const float* op = myobs + (size_t)t * 3;
            ax.v[0] = (half_t)((op[0] - nm0) * is0);
            ax.v[1] = (half_t)((op[1] - nm1) * is1);
            ax.v[2] = (half_t)((op[2] - nm2) * is2);
            ax.v[3] = (half_t)1.0f;
        }

        // ---- GEMM1: this wave's unit quarter, gate-by-gate (i, ig, fg, og) ----
#pragma unroll
        for (int g = 0; g < 4; ++g) {
            const int nt = g*4 + wv;
            v8f a = vsplat(0.0f);
            a = wmma16(ha0,  ld_frag(sm + WHF_OFF/2 + ((0*16 + nt)*32 + lane)*16), a);
            a = wmma16(ha1,  ld_frag(sm + WHF_OFF/2 + ((1*16 + nt)*32 + lane)*16), a);
            a = wmma16(ax.v, ld_frag(sm + WHF_OFF/2 + ((2*16 + nt)*32 + lane)*16), a);
            if (g == 0) {
#pragma unroll
                for (int r = 0; r < 8; ++r) ti[r] = ftanh(a[r]);
            } else if (g == 1) {
#pragma unroll
                for (int r = 0; r < 8; ++r) ti[r] *= fsigm(a[r]);
            } else if (g == 2) {
#pragma unroll
                for (int r = 0; r < 8; ++r) c[r] = c[r]*fsigm(a[r] + 1.0f) + ti[r];
            } else {
#pragma unroll
                for (int r = 0; r < 8; ++r) hl[r] = ftanh(c[r]) * fsigm(a[r]);
            }
        }

        // ---- h_lstm quarter -> shared LDS;全waves need full K=64 A fragments ----
#pragma unroll
        for (int r = 0; r < 8; ++r)
            hL_s[(r + 8*hsel)*64 + wv*16 + col] = (half_t)hl[r];
        __syncthreads();
        v16h la0 = ld_afrag(hL_s, 64, 0, m, hsel);
        v16h la1 = ld_afrag(hL_s, 64, 1, m, hsel);

        // ---- GEMM2: bb = lecun_tanh([h_lstm|obs|1] @ bb_aug); 2 n-tiles per wave ----
#pragma unroll
        for (int q = 0; q < 2; ++q) {
            const int nt = 2*wv + q;
            v8f a = vsplat(0.0f);
            a = wmma16(la0,  ld_frag(sm + BBF_OFF/2 + ((0*8 + nt)*32 + lane)*16), a);
            a = wmma16(la1,  ld_frag(sm + BBF_OFF/2 + ((1*8 + nt)*32 + lane)*16), a);
            a = wmma16(ax.v, ld_frag(sm + BBF_OFF/2 + ((2*8 + nt)*32 + lane)*16), a);
#pragma unroll
            for (int r = 0; r < 8; ++r)
                bb_s[(r + 8*hsel)*128 + nt*16 + col] = (half_t)(1.7159f * ftanh(0.666f * a[r]));
        }
        __syncthreads();
        v16h ba0 = ld_afrag(bb_s, 128, 0, m, hsel);
        v16h ba1 = ld_afrag(bb_s, 128, 1, m, hsel);
        v16h ba2 = ld_afrag(bb_s, 128, 2, m, hsel);
        v16h ba3 = ld_afrag(bb_s, 128, 3, m, hsel);

        // ---- GEMM3: [ff1|ff2|tc] n-tile wv;  h_new = ff1*(1-t)+t*ff2 ----
        v8f a1 = vsplat(fb1_r);
        a1 = wmma16(ba0, ld_frag(sm + FFF_OFF/2 + ((0*12 + wv)*32 + lane)*16), a1);
        a1 = wmma16(ba1, ld_frag(sm + FFF_OFF/2 + ((1*12 + wv)*32 + lane)*16), a1);
        a1 = wmma16(ba2, ld_frag(sm + FFF_OFF/2 + ((2*12 + wv)*32 + lane)*16), a1);
        a1 = wmma16(ba3, ld_frag(sm + FFF_OFF/2 + ((3*12 + wv)*32 + lane)*16), a1);
        v8f a2 = vsplat(fb2_r);
        a2 = wmma16(ba0, ld_frag(sm + FFF_OFF/2 + ((0*12 + 4 + wv)*32 + lane)*16), a2);
        a2 = wmma16(ba1, ld_frag(sm + FFF_OFF/2 + ((1*12 + 4 + wv)*32 + lane)*16), a2);
        a2 = wmma16(ba2, ld_frag(sm + FFF_OFF/2 + ((2*12 + 4 + wv)*32 + lane)*16), a2);
        a2 = wmma16(ba3, ld_frag(sm + FFF_OFF/2 + ((3*12 + 4 + wv)*32 + lane)*16), a2);
        v8f a3 = vsplat(fbt_r);
        a3 = wmma16(ba0, ld_frag(sm + FFF_OFF/2 + ((0*12 + 8 + wv)*32 + lane)*16), a3);
        a3 = wmma16(ba1, ld_frag(sm + FFF_OFF/2 + ((1*12 + 8 + wv)*32 + lane)*16), a3);
        a3 = wmma16(ba2, ld_frag(sm + FFF_OFF/2 + ((2*12 + 8 + wv)*32 + lane)*16), a3);
        a3 = wmma16(ba3, ld_frag(sm + FFF_OFF/2 + ((3*12 + 8 + wv)*32 + lane)*16), a3);
#pragma unroll
        for (int r = 0; r < 8; ++r) {
            float tv = fsigm(a3[r]);
            hn[r] = ftanh(a1[r]) * (1.0f - tv) + tv * ftanh(a2[r]);
        }

        // ---- h_new quarter -> shared LDS (separate buffer from hL_s) ----
#pragma unroll
        for (int r = 0; r < 8; ++r)
            hN_s[(r + 8*hsel)*64 + wv*16 + col] = (half_t)hn[r];
        if (t == T_SZ - 1) {
#pragma unroll
            for (int r = 0; r < 8; ++r) {
                const int row = brow + r + 8*hsel;
                out_hT[(size_t)row*64 + wv*16 + col] = hn[r];
                out_cT[(size_t)row*64 + wv*16 + col] = c[r];
            }
        }
        __syncthreads();
        ha0 = ld_afrag(hN_s, 64, 0, m, hsel);
        ha1 = ld_afrag(hN_s, 64, 1, m, hsel);

        // ---- proj (64->16) + action mean, wave 0 only ----
        if (wv == 0) {
            v8f p = vsplat(pjb_r);
            p = wmma16(ha0, ld_frag(sm + PJF_OFF/2 + (0*32 + lane)*16), p);
            p = wmma16(ha1, ld_frag(sm + PJF_OFF/2 + (1*32 + lane)*16), p);
#pragma unroll
            for (int r = 0; r < 8; ++r) {
                float v = p[r] * aw;
                v += __shfl_xor(v, 1, 16);
                v += __shfl_xor(v, 2, 16);
                v += __shfl_xor(v, 4, 16);
                v += __shfl_xor(v, 8, 16);
                if (col == 0) {
                    const int row = brow + r + 8*hsel;
                    out_mean[(size_t)row * T_SZ + t] = 2.0f * ftanh(v + amb);
                }
            }
        }
    }
}

// ===================== value MLP (obs->64->64->1) + log_std fill =====================
__global__ __launch_bounds__(128) void value_kernel(
    const float* __restrict__ obs,
    const float* __restrict__ v1_w, const float* __restrict__ v1_b,
    const float* __restrict__ v2_b, const float* __restrict__ v3_w, const float* __restrict__ v3_b,
    const float* __restrict__ als,
    const half_t* __restrict__ wsf,
    float* __restrict__ out_logstd, float* __restrict__ out_value)
{
    __shared__ half_t v2f_s[8*32*16];    // 8 fragments
    __shared__ float  w1_s[192], b1_s[64], b2_s[64], w3_s[64];
    const int tid  = threadIdx.x;
    const int lane = tid & 31;
    const int wv   = tid >> 5;

    for (int i = tid; i < (8*32*16)/8; i += blockDim.x)
        ((uint4*)v2f_s)[i] = ((const uint4*)(wsf + WS_V2F/2))[i];
    for (int i = tid; i < 192; i += blockDim.x) w1_s[i] = v1_w[i];
    for (int i = tid; i < 64;  i += blockDim.x) {
        b1_s[i] = v1_b[i];
        b2_s[i] = v2_b[i];
        w3_s[i] = v3_w[i];
    }
    __syncthreads();

    const int col  = lane & 15;
    const int hsel = lane >> 4;
    const size_t r0 = ((size_t)blockIdx.x * 4 + wv) * 16;

    // layer 1 (K=3) computed per-lane directly in A layout, with relu
    const float* op = obs + (r0 + col) * 3;
    const float o0 = op[0], o1 = op[1], o2 = op[2];
    V16 fa0, fa1;
#pragma unroll
    for (int kt = 0; kt < 2; ++kt) {
#pragma unroll
        for (int j = 0; j < 16; ++j) {
            const int u = kt*32 + hsel*16 + j;
            float v = b1_s[u] + o0*w1_s[u] + o1*w1_s[64 + u] + o2*w1_s[128 + u];
            v = fmaxf(v, 0.0f);
            if (kt == 0) fa0.v[j] = (half_t)v; else fa1.v[j] = (half_t)v;
        }
    }

    // layer 2 (64x64) via WMMA, relu
    v8f racc[4];
#pragma unroll
    for (int nt = 0; nt < 4; ++nt) {
        v8f a = vsplat(b2_s[nt*16 + col]);
        a = wmma16(fa0.v, ld_frag(v2f_s + ((0*4 + nt)*32 + lane)*16), a);
        a = wmma16(fa1.v, ld_frag(v2f_s + ((1*4 + nt)*32 + lane)*16), a);
#pragma unroll
        for (int r = 0; r < 8; ++r) a[r] = fmaxf(a[r], 0.0f);
        racc[nt] = a;
    }

    // layer 3 (64->1): per-row dot + half-wave shuffle reduction; also fill log_std
    const float vb = v3_b[0];
    const float ls = als[0];
#pragma unroll
    for (int r = 0; r < 8; ++r) {
        float s = racc[0][r]*w3_s[col] + racc[1][r]*w3_s[16 + col]
                + racc[2][r]*w3_s[32 + col] + racc[3][r]*w3_s[48 + col];
        s += __shfl_xor(s, 1, 16);
        s += __shfl_xor(s, 2, 16);
        s += __shfl_xor(s, 4, 16);
        s += __shfl_xor(s, 8, 16);
        if (col == 0) {
            const size_t row = r0 + r + 8*hsel;
            out_value[row]  = s + vb;
            out_logstd[row] = ls;
        }
    }
}

extern "C" void kernel_launch(void* const* d_in, const int* in_sizes, int n_in,
                              void* d_out, int out_size, void* d_ws, size_t ws_size,
                              hipStream_t stream) {
    (void)in_sizes; (void)n_in; (void)out_size; (void)ws_size;
    const float* obs      = (const float*)d_in[0];
    const float* rms_mean = (const float*)d_in[1];
    const float* rms_var  = (const float*)d_in[2];
    const float* lstm_wi  = (const float*)d_in[3];
    const float* lstm_bi  = (const float*)d_in[4];
    const float* lstm_wh  = (const float*)d_in[5];
    const float* bb_w     = (const float*)d_in[6];
    const float* bb_b     = (const float*)d_in[7];
    const float* ff1_w    = (const float*)d_in[8];
    const float* ff1_b    = (const float*)d_in[9];
    const float* ff2_w    = (const float*)d_in[10];
    const float* ff2_b    = (const float*)d_in[11];
    const float* ta_w     = (const float*)d_in[12];
    const float* ta_b     = (const float*)d_in[13];
    const float* tb_w     = (const float*)d_in[14];
    const float* tb_b     = (const float*)d_in[15];
    const float* proj_w   = (const float*)d_in[16];
    const float* proj_b   = (const float*)d_in[17];
    const float* am_w     = (const float*)d_in[18];
    const float* am_b     = (const float*)d_in[19];
    const float* als      = (const float*)d_in[20];
    const float* v1_w     = (const float*)d_in[21];
    const float* v1_b     = (const float*)d_in[22];
    const float* v2_w     = (const float*)d_in[23];
    const float* v2_b     = (const float*)d_in[24];
    const float* v3_w     = (const float*)d_in[25];
    const float* v3_b     = (const float*)d_in[26];

    half_t* ws = (half_t*)d_ws;
    float* out = (float*)d_out;
    float* out_mean   = out;
    float* out_logstd = out + (size_t)B_SZ * T_SZ;
    float* out_value  = out + 2 * (size_t)B_SZ * T_SZ;
    float* out_hT     = out + 3 * (size_t)B_SZ * T_SZ;
    float* out_cT     = out_hT + (size_t)B_SZ * 64;

    prep_kernel<<<64, 256, 0, stream>>>(lstm_wh, lstm_wi, lstm_bi, bb_w, bb_b,
                                        ff1_w, ff2_w, ta_w, tb_w, proj_w, v2_w, ws);
    ltc_kernel<<<B_SZ / 16, 128, LDS_TOTAL, stream>>>(
        obs, rms_mean, rms_var, ff1_b, ff2_b, ta_b, tb_b, proj_b, am_w, am_b,
        ws, out_mean, out_hT, out_cT);
    value_kernel<<<(B_SZ * T_SZ / 16) / 4, 128, 0, stream>>>(
        obs, v1_w, v1_b, v2_b, v3_w, v3_b, als, ws, out_logstd, out_value);
}
